// SpikeMixtralMoEBlock_5171140624985
// MI455X (gfx1250) — compile-verified, matrix operand order checked by reading, not run on hardware
//
#include <hip/hip_runtime.h>
#include <hip/hip_bf16.h>
#include <stdint.h>

// ---------------- problem constants (from reference) ----------------
#define T_TOK 4096   // B*S tokens
#define DDIM  2048
#define FDIM  4096
#define NEXP  8
#define TOPK  2

// ---------------- vector types ----------------
typedef __attribute__((ext_vector_type(16))) __bf16       v16bf;
typedef __attribute__((ext_vector_type(8)))  float        v8f;
typedef __attribute__((ext_vector_type(4)))  unsigned int u32x4;
typedef __attribute__((ext_vector_type(4)))  float        f32x4;
typedef __attribute__((ext_vector_type(4)))  int          i32x4;

union ABFrag {
  v16bf v;
  unsigned short s[16];
  u32x4 q[2];
};

__device__ __forceinline__ unsigned short f2bf(float f) {
  unsigned u = __builtin_bit_cast(unsigned, f);
  u += 0x7FFFu + ((u >> 16) & 1u);          // round-to-nearest-even
  return (unsigned short)(u >> 16);
}

#define WMMA_BF16(A, B, C) \
  __builtin_amdgcn_wmma_f32_16x16x32_bf16(false, (A).v, false, (B).v, (short)0, (C), false, false)

// ---------------- CDNA5 async global->LDS copy (ASYNCcnt path) ----------------
#if defined(__has_builtin)
#if __has_builtin(__builtin_amdgcn_global_load_async_to_lds_b128)
#define HAS_ASYNC_LDS 1
#endif
#if __has_builtin(__builtin_amdgcn_s_wait_asynccnt)
#define HAS_WAIT_ASYNC_BUILTIN 1
#endif
#endif

#ifndef HAS_ASYNC_LDS
#define HAS_ASYNC_LDS 0
#endif

#if HAS_ASYNC_LDS
// builtin is typed per payload width: b128 -> pointers to 4 x i32
typedef __attribute__((address_space(1))) i32x4* g1_v4i_ptr;   // global
typedef __attribute__((address_space(3))) i32x4* l3_v4i_ptr;   // LDS

__device__ __forceinline__ void async_copy16(const unsigned short* g, unsigned short* l) {
  // per-lane 16B global -> LDS, tracked with ASYNCcnt (GLOBAL_LOAD_ASYNC_TO_LDS_B128)
  __builtin_amdgcn_global_load_async_to_lds_b128(
      (g1_v4i_ptr)(uintptr_t)g,
      (l3_v4i_ptr)(unsigned int)(uintptr_t)l, 0, 0);
}

__device__ __forceinline__ void wait_async0() {
#if defined(HAS_WAIT_ASYNC_BUILTIN)
  __builtin_amdgcn_s_wait_asynccnt(0);
#else
  asm volatile("s_wait_asynccnt 0x0" ::: "memory");
#endif
}
#endif

// ---------------- workspace layout (bytes) ----------------
static constexpr size_t SZ_XB  = (size_t)T_TOK * DDIM * 2;          // 16 MB
static constexpr size_t SZ_W   = (size_t)NEXP * FDIM * DDIM * 2;    // 128 MB each
static constexpr size_t SZ_H   = (size_t)T_TOK * TOPK * FDIM * 2;   // 64 MB
static constexpr size_t SZ_YP  = (size_t)T_TOK * TOPK * DDIM * 4;   // 64 MB

static constexpr size_t OFF_XB  = 0;
static constexpr size_t OFF_W1B = OFF_XB  + SZ_XB;
static constexpr size_t OFF_W3B = OFF_W1B + SZ_W;
static constexpr size_t OFF_W2B = OFF_W3B + SZ_W;
static constexpr size_t OFF_H   = OFF_W2B + SZ_W;
static constexpr size_t OFF_YP  = OFF_H   + SZ_H;
static constexpr size_t OFF_PW  = OFF_YP  + SZ_YP;
static constexpr size_t OFF_EL  = OFF_PW  + (size_t)T_TOK * TOPK * 4;
static constexpr size_t OFF_CNT = OFF_EL  + (size_t)NEXP * T_TOK * 4;

// ---------------- kernel 0: router (one wave32 per token) ----------------
__global__ void moe_router(const float* __restrict__ x, const float* __restrict__ gw,
                           float* __restrict__ pair_wt, int* __restrict__ elist,
                           int* __restrict__ cnt) {
  const int wid  = threadIdx.x >> 5;
  const int lane = threadIdx.x & 31;
  const int t    = blockIdx.x * 8 + wid;

  float acc[NEXP];
#pragma unroll
  for (int e = 0; e < NEXP; ++e) acc[e] = 0.f;

  const float* xr = x + (size_t)t * DDIM;
  for (int d = lane; d < DDIM; d += 32) {
    const float xv = xr[d];
#pragma unroll
    for (int e = 0; e < NEXP; ++e) acc[e] += xv * gw[e * DDIM + d];
  }
#pragma unroll
  for (int e = 0; e < NEXP; ++e) {
#pragma unroll
    for (int off = 16; off > 0; off >>= 1) acc[e] += __shfl_xor(acc[e], off, 32);
  }

  if (lane == 0) {
    float m = acc[0];
#pragma unroll
    for (int e = 1; e < NEXP; ++e) m = fmaxf(m, acc[e]);
    float p[NEXP];
#pragma unroll
    for (int e = 0; e < NEXP; ++e) p[e] = __expf(acc[e] - m);
    int a0 = 0;
#pragma unroll
    for (int e = 1; e < NEXP; ++e) if (p[e] > p[a0]) a0 = e;
    int a1 = (a0 == 0) ? 1 : 0;
#pragma unroll
    for (int e = 0; e < NEXP; ++e) if (e != a0 && p[e] > p[a1]) a1 = e;
    const float den = p[a0] + p[a1];
    pair_wt[t * 2 + 0] = p[a0] / den;
    pair_wt[t * 2 + 1] = p[a1] / den;
    const int r0 = atomicAdd(&cnt[a0], 1);
    elist[a0 * T_TOK + r0] = t * 2 + 0;
    const int r1 = atomicAdd(&cnt[a1], 1);
    elist[a1 * T_TOK + r1] = t * 2 + 1;
  }
}

// ---------------- kernel 1: fp32 -> bf16 cast (x and all weights) ----------------
__global__ void cast_bf16(const float* __restrict__ src, unsigned short* __restrict__ dst,
                          int n8) {
  const int i = blockIdx.x * blockDim.x + threadIdx.x;   // 8 elements per thread
  if (i >= n8) return;
  const f32x4* p = (const f32x4*)src + (size_t)i * 2;
  const f32x4 f0 = p[0];
  const f32x4 f1 = p[1];
  u32x4 o;
  o[0] = (unsigned)f2bf(f0[0]) | ((unsigned)f2bf(f0[1]) << 16);
  o[1] = (unsigned)f2bf(f0[2]) | ((unsigned)f2bf(f0[3]) << 16);
  o[2] = (unsigned)f2bf(f1[0]) | ((unsigned)f2bf(f1[1]) << 16);
  o[3] = (unsigned)f2bf(f1[2]) | ((unsigned)f2bf(f1[3]) << 16);
  ((u32x4*)dst)[i] = o;
}

// B fragment (N x K 16x32 tile, per ISA 7.12.2: lane&15 = N row, (lane>>4)*8 = K chunk,
// elems 0..7 = K[kb..kb+7], elems 8..15 = K[kb+16..kb+23]). Row-major bf16 weights.
__device__ __forceinline__ ABFrag loadB_bf16(const unsigned short* __restrict__ p) {
  __builtin_prefetch(p + 32, 0, 3);                      // global_prefetch_b8, next K-step
  ABFrag b;
  b.q[0] = *(const u32x4*)p;
  b.q[1] = *(const u32x4*)(p + 16);
  return b;
}

// ---------------- kernel 2: per-expert SwiGLU up-projection ----------------
// Block tile: 64 gathered tokens x 128 of F. 8 waves (2M x 4N), each 32x32 out:
// per K-step 4 ds_load_b128 (A) + 8 global_load_b128 (B) + 8 v_wmma; the A tile
// for step k+1 is staged into the other LDS buffer with async-to-LDS copies.
__global__ __launch_bounds__(256) void moe_ffn1(
    const unsigned short* __restrict__ xb, const unsigned short* __restrict__ w1b,
    const unsigned short* __restrict__ w3b, const int* __restrict__ elist,
    const int* __restrict__ cnt, unsigned short* __restrict__ h_all) {
  __shared__ unsigned short As[2][64][40];   // double-buffered 64x32 bf16 (+8 pad)
  __shared__ int slotTile[64];

  const int e     = blockIdx.z;
  const int cntE  = cnt[e];
  const int mBase = blockIdx.y * 64;
  if (mBase >= cntE) return;              // block-uniform exit (EXEC stays full)

  const int tid = threadIdx.x, lane = tid & 31, wid = tid >> 5;
  if (tid < 64) {
    const int row = mBase + tid;
    slotTile[tid] = elist[e * T_TOK + (row < cntE ? row : 0)];
  }
  __syncthreads();

  const int waveM = wid >> 2, waveN = wid & 3;
  const int wm    = waveM * 32;
  const int nBase = blockIdx.x * 128 + waveN * 32;
  const unsigned short* W1 = w1b + (size_t)e * FDIM * DDIM;
  const unsigned short* W3 = w3b + (size_t)e * FDIM * DDIM;

  v8f c1[2][2] = {}, c3[2][2] = {};
  const int r0 = wm + (lane & 15), r1 = r0 + 16;
  const int kb = (lane >> 4) * 8;
  const size_t n0 = (size_t)(nBase + (lane & 15));
  const size_t n1 = n0 + 16;
  const int ar = tid >> 2, ac = (tid & 3) * 8;

  // hoisted per-thread A staging pointers (removes per-iter ds_load + addr rebuild)
  const unsigned short* aSrc = xb + (size_t)(slotTile[ar] >> 1) * DDIM + ac;

  constexpr int NK = DDIM / 32;
#if HAS_ASYNC_LDS
  async_copy16(aSrc, &As[0][ar][ac]);      // prologue: tile 0
#endif
  for (int kt = 0; kt < NK; ++kt) {
#if HAS_ASYNC_LDS
    wait_async0();                         // my lanes' part of tile kt landed
    __syncthreads();                       // everyone's part landed; buf kt-1 free
    if (kt + 1 < NK) async_copy16(aSrc + (kt + 1) * 32, &As[(kt + 1) & 1][ar][ac]);
#else
    __syncthreads();
    *(u32x4*)&As[kt & 1][ar][ac] = *(const u32x4*)(aSrc + kt * 32);
    __syncthreads();
#endif
    const unsigned short (*cur)[40] = As[kt & 1];
    const int k = kt * 32;

    ABFrag a0, a1;
    a0.q[0] = *(const u32x4*)&cur[r0][kb];  a0.q[1] = *(const u32x4*)&cur[r0][kb + 16];
    a1.q[0] = *(const u32x4*)&cur[r1][kb];  a1.q[1] = *(const u32x4*)&cur[r1][kb + 16];

    const ABFrag b1a = loadB_bf16(W1 + n0 * DDIM + k + kb);
    const ABFrag b1b = loadB_bf16(W1 + n1 * DDIM + k + kb);
    const ABFrag b3a = loadB_bf16(W3 + n0 * DDIM + k + kb);
    const ABFrag b3b = loadB_bf16(W3 + n1 * DDIM + k + kb);

    c1[0][0] = WMMA_BF16(a0, b1a, c1[0][0]);
    c1[0][1] = WMMA_BF16(a0, b1b, c1[0][1]);
    c1[1][0] = WMMA_BF16(a1, b1a, c1[1][0]);
    c1[1][1] = WMMA_BF16(a1, b1b, c1[1][1]);
    c3[0][0] = WMMA_BF16(a0, b3a, c3[0][0]);
    c3[0][1] = WMMA_BF16(a0, b3b, c3[0][1]);
    c3[1][0] = WMMA_BF16(a1, b3a, c3[1][0]);
    c3[1][1] = WMMA_BF16(a1, b3b, c3[1][1]);
  }

  // SwiGLU epilogue; C/D layout: elem r -> M = r + 8*(lane>>4), N = lane&15
#pragma unroll
  for (int mi = 0; mi < 2; ++mi) {
#pragma unroll
    for (int ni = 0; ni < 2; ++ni) {
      const int n = nBase + ni * 16 + (lane & 15);
#pragma unroll
      for (int r = 0; r < 8; ++r) {
        const int lr = wm + mi * 16 + 8 * (lane >> 4) + r;
        if (mBase + lr < cntE) {
          const float z  = c1[mi][ni][r];
          const float hv = (z / (1.f + __expf(-z))) * c3[mi][ni][r];
          h_all[(size_t)slotTile[lr] * FDIM + n] = f2bf(hv);
        }
      }
    }
  }
}

// ---------------- kernel 3: per-expert down-projection ----------------
__global__ __launch_bounds__(256) void moe_ffn2(
    const unsigned short* __restrict__ h_all, const unsigned short* __restrict__ w2b,
    const int* __restrict__ elist, const int* __restrict__ cnt,
    const float* __restrict__ pair_wt, float* __restrict__ yp) {
  __shared__ unsigned short As[2][64][40];
  __shared__ int   slotTile[64];
  __shared__ float wtTile[64];

  const int e     = blockIdx.z;
  const int cntE  = cnt[e];
  const int mBase = blockIdx.y * 64;
  if (mBase >= cntE) return;

  const int tid = threadIdx.x, lane = tid & 31, wid = tid >> 5;
  if (tid < 64) {
    const int row = mBase + tid;
    const int s   = elist[e * T_TOK + (row < cntE ? row : 0)];
    slotTile[tid] = s;
    wtTile[tid]   = pair_wt[s];
  }
  __syncthreads();

  const int waveM = wid >> 2, waveN = wid & 3;
  const int wm    = waveM * 32;
  const int nBase = blockIdx.x * 128 + waveN * 32;
  const unsigned short* W2 = w2b + (size_t)e * DDIM * FDIM;

  v8f c[2][2] = {};
  const int r0 = wm + (lane & 15), r1 = r0 + 16;
  const int kb = (lane >> 4) * 8;
  const size_t n0 = (size_t)(nBase + (lane & 15));
  const size_t n1 = n0 + 16;
  const int ar = tid >> 2, ac = (tid & 3) * 8;

  const unsigned short* aSrc = h_all + (size_t)slotTile[ar] * FDIM + ac;

  constexpr int NK = FDIM / 32;
#if HAS_ASYNC_LDS
  async_copy16(aSrc, &As[0][ar][ac]);
#endif
  for (int kt = 0; kt < NK; ++kt) {
#if HAS_ASYNC_LDS
    wait_async0();
    __syncthreads();
    if (kt + 1 < NK) async_copy16(aSrc + (kt + 1) * 32, &As[(kt + 1) & 1][ar][ac]);
#else
    __syncthreads();
    *(u32x4*)&As[kt & 1][ar][ac] = *(const u32x4*)(aSrc + kt * 32);
    __syncthreads();
#endif
    const unsigned short (*cur)[40] = As[kt & 1];
    const int k = kt * 32;

    ABFrag a0, a1;
    a0.q[0] = *(const u32x4*)&cur[r0][kb];  a0.q[1] = *(const u32x4*)&cur[r0][kb + 16];
    a1.q[0] = *(const u32x4*)&cur[r1][kb];  a1.q[1] = *(const u32x4*)&cur[r1][kb + 16];

    const ABFrag b0 = loadB_bf16(W2 + n0 * FDIM + k + kb);
    const ABFrag b1 = loadB_bf16(W2 + n1 * FDIM + k + kb);

    c[0][0] = WMMA_BF16(a0, b0, c[0][0]);
    c[0][1] = WMMA_BF16(a0, b1, c[0][1]);
    c[1][0] = WMMA_BF16(a1, b0, c[1][0]);
    c[1][1] = WMMA_BF16(a1, b1, c[1][1]);
  }

#pragma unroll
  for (int mi = 0; mi < 2; ++mi) {
#pragma unroll
    for (int ni = 0; ni < 2; ++ni) {
      const int n = nBase + ni * 16 + (lane & 15);
#pragma unroll
      for (int r = 0; r < 8; ++r) {
        const int lr = wm + mi * 16 + 8 * (lane >> 4) + r;
        if (mBase + lr < cntE) {
          yp[(size_t)slotTile[lr] * DDIM + n] = wtTile[lr] * c[mi][ni][r];
        }
      }
    }
  }
}

// ---------------- kernel 4: combine the two routed contributions ----------------
__global__ void moe_combine(const float* __restrict__ yp, float* __restrict__ y) {
  const int i = blockIdx.x * blockDim.x + threadIdx.x;  // over T*D/4 float4s
  const int t  = i / (DDIM / 4);
  const int d4 = i % (DDIM / 4);
  const f32x4 a = ((const f32x4*)yp)[(size_t)(2 * t + 0) * (DDIM / 4) + d4];
  const f32x4 b = ((const f32x4*)yp)[(size_t)(2 * t + 1) * (DDIM / 4) + d4];
  ((f32x4*)y)[i] = a + b;
}

// ---------------- launcher ----------------
extern "C" void kernel_launch(void* const* d_in, const int* in_sizes, int n_in,
                              void* d_out, int out_size, void* d_ws, size_t ws_size,
                              hipStream_t stream) {
  (void)in_sizes; (void)n_in; (void)out_size; (void)ws_size;
  const float* x  = (const float*)d_in[0];   // [B,S,D]
  const float* gw = (const float*)d_in[1];   // [E,D]
  const float* w1 = (const float*)d_in[2];   // [E,F,D]
  const float* w2 = (const float*)d_in[3];   // [E,D,F]
  const float* w3 = (const float*)d_in[4];   // [E,F,D]
  float* y = (float*)d_out;

  char* ws = (char*)d_ws;
  unsigned short* xb      = (unsigned short*)(ws + OFF_XB);
  unsigned short* w1b     = (unsigned short*)(ws + OFF_W1B);
  unsigned short* w3b     = (unsigned short*)(ws + OFF_W3B);
  unsigned short* w2b     = (unsigned short*)(ws + OFF_W2B);
  unsigned short* h_all   = (unsigned short*)(ws + OFF_H);
  float*          yp      = (float*)(ws + OFF_YP);
  float*          pair_wt = (float*)(ws + OFF_PW);
  int*            elist   = (int*)(ws + OFF_EL);
  int*            cnt     = (int*)(ws + OFF_CNT);

  (void)hipMemsetAsync(cnt, 0, NEXP * sizeof(int), stream);

  moe_router<<<T_TOK / 8, 256, 0, stream>>>(x, gw, pair_wt, elist, cnt);

  const int nx8 = T_TOK * DDIM / 8;              // x elements / 8
  const int nw8 = NEXP * FDIM * DDIM / 8;        // one weight tensor / 8
  cast_bf16<<<(nx8 + 255) / 256, 256, 0, stream>>>(x, xb, nx8);
  cast_bf16<<<(nw8 + 255) / 256, 256, 0, stream>>>(w1, w1b, nw8);
  cast_bf16<<<(nw8 + 255) / 256, 256, 0, stream>>>(w3, w3b, nw8);
  cast_bf16<<<(nw8 + 255) / 256, 256, 0, stream>>>(w2, w2b, nw8);

  moe_ffn1<<<dim3(FDIM / 128, T_TOK / 64, NEXP), 256, 0, stream>>>(xb, w1b, w3b, elist, cnt, h_all);
  moe_ffn2<<<dim3(DDIM / 128, T_TOK / 64, NEXP), 256, 0, stream>>>(h_all, w2b, elist, cnt, pair_wt, yp);
  moe_combine<<<(T_TOK * DDIM / 4) / 256, 256, 0, stream>>>(yp, y);
}
